// SA_upsample_48223892800142
// MI455X (gfx1250) — matrix operand order repaired and checked by reading, not compiled, not run on hardware
//
#include <hip/hip_runtime.h>
#include <hip/hip_bf16.h>
#include <math.h>

// ---------------- problem constants (reference: B,C,E,H,W = 8,64,2,128,128; scale=2) ---
#define BB 8
#define CC 64
#define HH 128
#define WW 128
#define HS 256
#define WS 256
#define PP (HS * WS)          // 65536 output pixels

typedef __attribute__((ext_vector_type(16))) _Float16 v16h;
typedef __attribute__((ext_vector_type(8)))  float    v8f;

// =====================================================================================
// Kernel 0: build fixed f16 A-fragments for the two WMMA stages.
//   Stage-2 A = Wcomb (16 x 64):  row m = weight_compress[m/8][m%8][:], split K into 2x32
//   Stage-4 A_m (m=0..3) = Wexp rows [16m..16m+16) x 16 (K padded to 32 with zeros)
//   A-fragment layout used here (16-bit A 16x32, contiguous K, same as B): lane = M + 16*hi,
//   half h -> K = h + 16*hi
// =====================================================================================
__global__ void build_frags_kernel(const float* __restrict__ wc,   // (2,8,64)
                                   const float* __restrict__ we,   // (2,64,8)
                                   _Float16* __restrict__ fragA2,  // [2][32][16]
                                   _Float16* __restrict__ fragA4)  // [4][32][16]
{
    int lane = threadIdx.x;        // 32 threads
    int m  = lane & 15;
    int hi = lane >> 4;
    #pragma unroll
    for (int f = 0; f < 2; ++f) {
        #pragma unroll
        for (int h = 0; h < 16; ++h) {
            int k = h + 16 * hi;
            int c = k + 32 * f;                       // contraction channel
            // Wcomb[m][c] = wc[m*64 + c]   (row m == (e= m>>3, k= m&7))
            fragA2[(f * 32 + lane) * 16 + h] = (_Float16)wc[m * 64 + c];
        }
    }
    #pragma unroll
    for (int mm = 0; mm < 4; ++mm) {
        #pragma unroll
        for (int h = 0; h < 16; ++h) {
            int k = h + 16 * hi;
            int c = mm * 16 + m;                      // output channel row
            float v = 0.f;
            if (k < 16) v = we[(k >> 3) * CC * 8 + c * 8 + (k & 7)]; // Wexp[c][k]
            fragA4[(mm * 32 + lane) * 16 + h] = (_Float16)v;
        }
    }
}

// =====================================================================================
// Kernel 1: per-pixel MLP -> routing (r0,r1) + bilinear sampling metadata.
// Batch independent: P = 65536 pixels.
// =====================================================================================
__global__ void __launch_bounds__(256) precompute_kernel(
    const float* __restrict__ w1, const float* __restrict__ b1,
    const float* __restrict__ w2, const float* __restrict__ b2,
    const float* __restrict__ w_r, const float* __restrict__ b_r,
    const float* __restrict__ w_o, const float* __restrict__ b_o,
    const int* __restrict__ scale_p, const int* __restrict__ scale2_p,
    float4* __restrict__ meta4, int2* __restrict__ metai)
{
    int p = blockIdx.x * blockDim.x + threadIdx.x;
    if (p >= PP) return;
    int iy = p / WS, ix = p % WS;
    float scale  = (float)scale_p[0];
    float scale2 = (float)scale2_p[0];

    float chv = (iy + 0.5f) / scale;
    float coor_h = chv - floorf(chv + 0.001f) - 0.5f;
    float cwv = (ix + 0.5f) / scale2;
    float coor_w = cwv - floorf(cwv + 0.001f) - 0.5f;
    float i0 = 1.f / scale2, i1 = 1.f / scale;

    float e1[32];
    #pragma unroll
    for (int o = 0; o < 32; ++o) {
        float s = b1[o];
        s = fmaf(w1[o * 4 + 0], i0, s);
        s = fmaf(w1[o * 4 + 1], i1, s);
        s = fmaf(w1[o * 4 + 2], coor_h, s);
        s = fmaf(w1[o * 4 + 3], coor_w, s);
        e1[o] = fmaxf(s, 0.f);
    }
    float e2[32];
    #pragma unroll
    for (int o = 0; o < 32; ++o) {
        float s = b2[o];
        #pragma unroll
        for (int i = 0; i < 32; ++i) s = fmaf(w2[o * 32 + i], e1[i], s);
        e2[o] = fmaxf(s, 0.f);
    }
    float r[2], off[2];
    #pragma unroll
    for (int e = 0; e < 2; ++e) {
        float s = b_r[e];
        #pragma unroll
        for (int i = 0; i < 32; ++i) s = fmaf(w_r[e * 32 + i], e2[i], s);
        r[e] = 1.f / (1.f + __expf(-s));
        float t = b_o[e];
        #pragma unroll
        for (int i = 0; i < 32; ++i) t = fmaf(w_o[e * 32 + i], e2[i], t);
        off[e] = t;
    }
    // sample position (simplified: px = (ix+0.5)/scale2 - 0.5 + off0)
    float pxf = (ix + 0.5f) / scale2 - 0.5f + off[0];
    float pyf = (iy + 0.5f) / scale  - 0.5f + off[1];
    float x0 = floorf(pxf), y0 = floorf(pyf);
    meta4[p] = make_float4(r[0], r[1], pxf - x0, pyf - y0);
    metai[p] = make_int2((int)x0, (int)y0);
}

// =====================================================================================
// Kernel 2: fused gather + two WMMA stages + output.
// Block = 128 threads = 4 waves; each wave owns one 16-pixel tile, loops over b=0..7.
// =====================================================================================
__global__ void __launch_bounds__(128) sa_main_kernel(
    const float* __restrict__ x,
    const float4* __restrict__ meta4, const int2* __restrict__ metai,
    const _Float16* __restrict__ fragA2, const _Float16* __restrict__ fragA4,
    float* __restrict__ out)
{
    __shared__ float feaLds[4][64 * 17];   // [wave][c*17 + px], stride 17 vs 64 banks

    const int wave = threadIdx.x >> 5;
    const int lane = threadIdx.x & 31;
    const int hi   = lane >> 4;            // 0 for lanes 0-15, 1 for 16-31
    const int px   = lane & 15;            // pixel column within tile
    const int p0   = (blockIdx.x * 4 + wave) * 16;
    float* fea = feaLds[wave];

    // preload fixed A fragments (32B contiguous per lane)
    v16h a2[2], a4[4];
    a2[0] = *(const v16h*)(fragA2 + (0 * 32 + lane) * 16);
    a2[1] = *(const v16h*)(fragA2 + (1 * 32 + lane) * 16);
    #pragma unroll
    for (int m = 0; m < 4; ++m)
        a4[m] = *(const v16h*)(fragA4 + (m * 32 + lane) * 16);

    // per-pixel metadata (lanes 16-31 mirror lanes 0-15)
    float4 mt = meta4[p0 + px];
    int2   mi = metai[p0 + px];
    const float r0 = mt.x, r1 = mt.y, wx = mt.z, wy = mt.w;
    const int x0 = mi.x, y0 = mi.y;
    const float w00 = (1.f - wx) * (1.f - wy), w10 = wx * (1.f - wy);
    const float w01 = (1.f - wx) * wy,         w11 = wx * wy;
    const bool vx0 = (x0 >= 0) && (x0 < WW),     vx1 = (x0 + 1 >= 0) && (x0 + 1 < WW);
    const bool vy0 = (y0 >= 0) && (y0 < HH),     vy1 = (y0 + 1 >= 0) && (y0 + 1 < HH);
    const float f00 = (vx0 && vy0) ? w00 : 0.f,  f10 = (vx1 && vy0) ? w10 : 0.f;
    const float f01 = (vx0 && vy1) ? w01 : 0.f,  f11 = (vx1 && vy1) ? w11 : 0.f;
    const int cx0 = min(max(x0, 0), WW - 1),     cx1 = min(max(x0 + 1, 0), WW - 1);
    const int cy0 = min(max(y0, 0), HH - 1),     cy1 = min(max(y0 + 1, 0), HH - 1);
    const int i00 = cy0 * WW + cx0, i10 = cy0 * WW + cx1;
    const int i01 = cy1 * WW + cx0, i11 = cy1 * WW + cx1;
    const float g = hi ? 0.f : 1.f;    // stage-4 B fragment: K>=16 rows are zero

    for (int b = 0; b < BB; ++b) {
        const float* xb = x + (size_t)b * CC * HH * WW;
        // ---- bilinear gather: 64 ch x 16 px into LDS (f32) ----
        #pragma unroll 4
        for (int i = 0; i < 32; ++i) {
            int c = 2 * i + hi;
            const float* xc = xb + c * (HH * WW);
            float v = f00 * xc[i00] + f10 * xc[i10] + f01 * xc[i01] + f11 * xc[i11];
            fea[c * 17 + px] = v;          // same-wave DS ops are in-order
        }
        // ---- stage 2: mid_all(16 x 16px) = Wcomb(16x64) @ fea(64x16px) ----
        // B layout (16-bit 32x16): lane = n + 16*(k>=16), half = k&15
        v16h bf0, bf1;
        #pragma unroll
        for (int h = 0; h < 16; ++h) {
            bf0[h] = (_Float16)fea[(h + 16 * hi) * 17 + px];
            bf1[h] = (_Float16)fea[(h + 16 * hi + 32) * 17 + px];
        }
        v8f d2;
        #pragma unroll
        for (int v = 0; v < 8; ++v) d2[v] = 0.f;
        d2 = __builtin_amdgcn_wmma_f32_16x16x32_f16(false, a2[0], false, bf0,
                                                    (short)0, d2, false, false);
        d2 = __builtin_amdgcn_wmma_f32_16x16x32_f16(false, a2[1], false, bf1,
                                                    (short)0, d2, false, false);
        // ---- expert combine + routing scale -> stage-4 B fragment t(16x16px) ----
        // D row v(+8*hi): rows 0-7 = expert0 partials, 8-15 = expert1 partials
        v16h tb;
        #pragma unroll
        for (int v = 0; v < 8; ++v) {
            float other = __shfl_xor(d2[v], 16, 32);
            float m0v = hi ? other : d2[v];
            float m1v = hi ? d2[v] : other;
            float mid = fmaf(r0, m0v, r1 * m1v);       // mid = Wc[p] @ fea
            tb[v]     = (_Float16)(g * (r0 * mid));    // t rows 0-7
            tb[v + 8] = (_Float16)(g * (r1 * mid));    // t rows 8-15
        }
        // ---- stage 4: out_tile = Wexp_chunk(16x16) @ t + fea0_tile ----
        #pragma unroll
        for (int m = 0; m < 4; ++m) {
            v8f cacc;
            #pragma unroll
            for (int v = 0; v < 8; ++v)
                cacc[v] = fea[(m * 16 + v + 8 * hi) * 17 + px];   // C = fea0
            v8f d4 = __builtin_amdgcn_wmma_f32_16x16x32_f16(false, a4[m], false, tb,
                                                            (short)0, cacc, false, false);
            float* ob = out + ((size_t)(b * CC + m * 16 + 8 * hi)) * PP + p0 + px;
            #pragma unroll
            for (int v = 0; v < 8; ++v) ob[(size_t)v * PP] = d4[v];
        }
    }
}

// =====================================================================================
extern "C" void kernel_launch(void* const* d_in, const int* in_sizes, int n_in,
                              void* d_out, int out_size, void* d_ws, size_t ws_size,
                              hipStream_t stream) {
    const float* x    = (const float*)d_in[0];
    const float* wc   = (const float*)d_in[1];   // weight_compress (2,8,64)
    const float* we   = (const float*)d_in[2];   // weight_expand   (2,64,8)
    const float* w1   = (const float*)d_in[3];
    const float* b1   = (const float*)d_in[4];
    const float* w2   = (const float*)d_in[5];
    const float* b2   = (const float*)d_in[6];
    const float* w_r  = (const float*)d_in[7];
    const float* b_r  = (const float*)d_in[8];
    const float* w_o  = (const float*)d_in[9];
    const float* b_o  = (const float*)d_in[10];
    const int* scale  = (const int*)d_in[11];
    const int* scale2 = (const int*)d_in[12];
    float* out = (float*)d_out;

    // workspace layout: meta4 (P*16B) | metai (P*8B) | fragA2 (2KB f16) | fragA4 (4KB)
    char* ws = (char*)d_ws;
    float4*    meta4  = (float4*)ws;
    int2*      metai  = (int2*)(ws + (size_t)PP * sizeof(float4));
    _Float16*  fragA2 = (_Float16*)(ws + (size_t)PP * (sizeof(float4) + sizeof(int2)));
    _Float16*  fragA4 = fragA2 + 2 * 32 * 16;

    build_frags_kernel<<<1, 32, 0, stream>>>(wc, we, fragA2, fragA4);
    precompute_kernel<<<PP / 256, 256, 0, stream>>>(w1, b1, w2, b2, w_r, b_r,
                                                    w_o, b_o, scale, scale2,
                                                    meta4, metai);
    sa_main_kernel<<<PP / 64, 128, 0, stream>>>(x, meta4, metai, fragA2, fragA4, out);
}
